// MLPWrapper_17162689315242
// MI455X (gfx1250) — matrix operand order repaired and checked by reading, not compiled, not run on hardware
//
#include <hip/hip_runtime.h>
#include <hip/hip_bf16.h>
#include <math.h>

typedef __attribute__((ext_vector_type(8)))  float  v8f;
typedef __attribute__((ext_vector_type(16))) __bf16 v16bf;
typedef __attribute__((ext_vector_type(8)))  __bf16 v8bf;

#define HDIM 2048
#define IDIM 5504
#define NTOK 16384
#define EPSV 1e-6f

// ---------------- workspace layout (bytes) ----------------
#define OFF_CNT 0
#define OFF_IDX 256
#define W_DWORDS 5636096                     // (2048/2)*5504 == (5504/2)*2048
#define W_BYTES  (W_DWORDS * 4)
#define OFF_WGH (256 + 65536)
#define OFF_WGL (OFF_WGH + W_BYTES)
#define OFF_WUH (OFF_WGL + W_BYTES)
#define OFF_WUL (OFF_WUH + W_BYTES)
#define OFF_WDH (OFF_WUL + W_BYTES)
#define OFF_WDL (OFF_WDH + W_BYTES)

static __device__ __forceinline__ unsigned int pack_bf(__bf16 a, __bf16 b) {
  unsigned short ua = __builtin_bit_cast(unsigned short, a);
  unsigned short ub = __builtin_bit_cast(unsigned short, b);
  return (unsigned int)ua | ((unsigned int)ub << 16);
}

static __device__ __forceinline__ v8f wmma_bf16(v16bf a, v16bf b, v8f c) {
  return __builtin_amdgcn_wmma_f32_16x16x32_bf16(false, a, false, b, (short)0, c,
                                                 false, false);
}

// A-fragment (16x32 bf16) from LDS row-major tile.
// lanes 0-15: K = k0+0..15 ; lanes 16-31: K = k0+16..31
static __device__ __forceinline__ v16bf load_afrag(const __bf16* rowbase, int k0,
                                                   int half) {
  const v8bf* p0 = (const v8bf*)(rowbase + k0 + 16 * half);
  const v8bf* p1 = (const v8bf*)(rowbase + k0 + 16 * half + 8);
  v8bf x0 = *p0, x1 = *p1;
  return __builtin_shufflevector(x0, x1, 0, 1, 2, 3, 4, 5, 6, 7, 8, 9, 10, 11, 12,
                                 13, 14, 15);
}

// ---------------- kernel 0: zero masked-out rows ----------------
__global__ void init_out_kernel(const int* __restrict__ mask,
                                float* __restrict__ out) {
  int row = blockIdx.x;
  if (mask[row]) return;  // kept rows fully written by main kernel
  float4 z = make_float4(0.f, 0.f, 0.f, 0.f);
  float4* o = (float4*)(out + (size_t)row * HDIM);
  o[threadIdx.x] = z;
  o[threadIdx.x + 256] = z;
}

// ---------------- kernel 1: deterministic ordered compaction ----------------
__global__ void compact_kernel(const int* __restrict__ mask, int* __restrict__ idx,
                               int* __restrict__ cnt) {
  __shared__ int sc[1024];
  __shared__ int base;
  int tid = threadIdx.x;
  if (tid == 0) base = 0;
  __syncthreads();
  for (int chunk = 0; chunk < NTOK; chunk += 1024) {
    int i = chunk + tid;
    int v = mask[i] ? 1 : 0;
    sc[tid] = v;
    __syncthreads();
    for (int off = 1; off < 1024; off <<= 1) {
      int t = (tid >= off) ? sc[tid - off] : 0;
      __syncthreads();
      sc[tid] += t;
      __syncthreads();
    }
    if (v) idx[base + sc[tid] - 1] = i;
    __syncthreads();
    if (tid == 0) base += sc[1023];
    __syncthreads();
  }
  if (tid == 0) *cnt = base;
}

// ---------------- kernel 2: weight fp32 -> (hi,lo) bf16, B-fragment swizzle ----
// swizzled dword index: ((((ntile*(K/32) + kblk)*2 + half)*16 + nlo)*8 + j
// dword j holds {W[k], W[k+1]} (low half = even k), k = kblk*32 + half*16 + 2j
__global__ void convert_swz_kernel(const float* __restrict__ W,
                                   unsigned int* __restrict__ hi,
                                   unsigned int* __restrict__ lo, int K, int N) {
  size_t tid = (size_t)blockIdx.x * blockDim.x + threadIdx.x;
  size_t total = (size_t)(K / 2) * N;
  if (tid >= total) return;
  int n = (int)(tid % N);
  size_t k2 = tid / N;
  int k = (int)(k2 * 2);
  float a = W[(size_t)k * N + n];
  float b = W[(size_t)(k + 1) * N + n];
  __bf16 ah = (__bf16)a, bh = (__bf16)b;
  __bf16 al = (__bf16)(a - (float)ah), bl = (__bf16)(b - (float)bh);
  int ntile = n >> 4, nl = n & 15, kblk = k >> 5, hf = (k >> 4) & 1;
  int j = ((int)k2) & 7;
  size_t dst = ((((size_t)ntile * (K / 32) + kblk) * 2 + hf) * 16 + nl) * 8 + j;
  hi[dst] = pack_bf(ah, bh);
  lo[dst] = pack_bf(al, bl);
}

// ---------------- kernel 3: fused RMSNorm + GLU + down projection ------------
__global__ void __launch_bounds__(256, 1)
mlp_main_kernel(const float* __restrict__ hs, const float* __restrict__ lnw,
                const float* __restrict__ gate_b, const float* __restrict__ up_b,
                const float* __restrict__ down_b, const int* __restrict__ idx,
                const int* __restrict__ cntp,
                const unsigned int* __restrict__ wg_hi,
                const unsigned int* __restrict__ wg_lo,
                const unsigned int* __restrict__ wu_hi,
                const unsigned int* __restrict__ wu_lo,
                const unsigned int* __restrict__ wd_hi,
                const unsigned int* __restrict__ wd_lo,
                float* __restrict__ out) {
  __shared__ __align__(16) __bf16 h_hi[16][2056];   // padded stride vs 64 banks
  __shared__ __align__(16) __bf16 h_lo[16][2056];
  __shared__ __align__(16) __bf16 a_hi[16][136];
  __shared__ __align__(16) __bf16 a_lo[16][136];
  __shared__ float red[256];
  __shared__ float rstd_s[16];

  int cnt = *cntp;
  int tilebase = blockIdx.x * 16;
  if (tilebase >= cnt) return;  // uniform exit

  int w = threadIdx.x >> 5;      // wave id 0..7
  int ln = threadIdx.x & 31;     // lane
  int half = ln >> 4;
  int nlo = ln & 15;

  // ---- phase 1: load rows, RMSNorm, bf16 hi/lo split into LDS ----
  {
    int r = threadIdx.x >> 4, c = threadIdx.x & 15;
    int p = tilebase + r;
    bool vrow = p < cnt;
    const float* xb = vrow ? (hs + (size_t)idx[p] * HDIM) : hs;
    float s = 0.f;
    if (vrow) {
      for (int i = 0; i < 128; ++i) {
        float v = xb[c + 16 * i];
        s += v * v;
      }
    }
    red[threadIdx.x] = s;
    __syncthreads();
    if (threadIdx.x < 16) {
      float t = 0.f;
      for (int i = 0; i < 16; ++i) t += red[threadIdx.x * 16 + i];
      rstd_s[threadIdx.x] = rsqrtf(t * (1.0f / HDIM) + EPSV);
    }
    __syncthreads();
    float rs = rstd_s[r];
    for (int i = 0; i < 128; ++i) {
      int k = c + 16 * i;
      float hv = vrow ? (xb[k] * rs * lnw[k]) : 0.0f;
      __bf16 hh = (__bf16)hv;
      __bf16 hl = (__bf16)(hv - (float)hh);
      h_hi[r][k] = hh;
      h_lo[r][k] = hl;
    }
    __syncthreads();
  }

  // down-projection accumulators: wave w owns out cols [256w, 256w+256)
  v8f acc[16];
#pragma unroll
  for (int t = 0; t < 16; ++t) acc[t] = (v8f){};

  const __bf16* ah_row = &h_hi[nlo][0];
  const __bf16* al_row = &h_lo[nlo][0];
  const __bf16* dah_row = &a_hi[nlo][0];
  const __bf16* dal_row = &a_lo[nlo][0];

  for (int ic = 0; ic < 43; ++ic) {
    int i0 = ic * 128;
    int c0 = i0 + 16 * w;  // this wave's gate/up column tile
    v8f g = (v8f){}, u = (v8f){};
    size_t ct = (size_t)(c0 >> 4);
    const unsigned int* pgh = wg_hi + (((ct * 64) * 2 + half) * 16 + nlo) * 8;
    const unsigned int* pgl = wg_lo + (((ct * 64) * 2 + half) * 16 + nlo) * 8;
    const unsigned int* puh = wu_hi + (((ct * 64) * 2 + half) * 16 + nlo) * 8;
    const unsigned int* pul = wu_lo + (((ct * 64) * 2 + half) * 16 + nlo) * 8;

    for (int ks = 0; ks < 64; ++ks) {
      int k0 = ks * 32;
      v16bf Ah = load_afrag(ah_row, k0, half);
      v16bf Al = load_afrag(al_row, k0, half);
      v16bf Bgh = *(const v16bf*)pgh;
      v16bf Bgl = *(const v16bf*)pgl;
      v16bf Buh = *(const v16bf*)puh;
      v16bf Bul = *(const v16bf*)pul;
      // WGP-scope prefetch of next iteration's fragments (per-lane addresses
      // cover the full 1KB/stream consumed next step); hides L2 latency.
      __builtin_prefetch((const void*)(pgh + 256), 0, 3);
      __builtin_prefetch((const void*)(pgl + 256), 0, 3);
      __builtin_prefetch((const void*)(puh + 256), 0, 3);
      __builtin_prefetch((const void*)(pul + 256), 0, 3);
      g = wmma_bf16(Ah, Bgh, g);
      g = wmma_bf16(Al, Bgh, g);
      g = wmma_bf16(Ah, Bgl, g);
      u = wmma_bf16(Ah, Buh, u);
      u = wmma_bf16(Al, Buh, u);
      u = wmma_bf16(Ah, Bul, u);
      pgh += 256; pgl += 256; puh += 256; pul += 256;
    }

    float gb = gate_b[c0 + nlo];
    float ub = up_b[c0 + nlo];
    __syncthreads();  // previous iteration's act reads complete
#pragma unroll
    for (int j = 0; j < 8; ++j) {
      float gg = g[j] + gb;
      float uu = u[j] + ub;
      float sig = 1.0f / (1.0f + __expf(-gg));
      float a = gg * sig * uu;  // silu(g) * u
      __bf16 aH = (__bf16)a;
      __bf16 aL = (__bf16)(a - (float)aH);
      int m = j + 8 * half;
      a_hi[m][16 * w + nlo] = aH;
      a_lo[m][16 * w + nlo] = aL;
    }
    __syncthreads();

    // down projection: out[16,2048] += act[16,128] @ Wd[128,2048]
#pragma unroll
    for (int ks = 0; ks < 4; ++ks) {
      v16bf Ah = load_afrag(dah_row, ks * 32, half);
      v16bf Al = load_afrag(dal_row, ks * 32, half);
      size_t kb = (size_t)(ic * 4 + ks);
#pragma unroll
      for (int t = 0; t < 16; ++t) {
        size_t cct = (size_t)(16 * w + t);
        size_t off = (((cct * 172 + kb) * 2 + half) * 16 + nlo) * 8;
        v16bf Bh = *(const v16bf*)(wd_hi + off);
        v16bf Bl = *(const v16bf*)(wd_lo + off);
        acc[t] = wmma_bf16(Ah, Bh, acc[t]);
        acc[t] = wmma_bf16(Al, Bh, acc[t]);
        acc[t] = wmma_bf16(Ah, Bl, acc[t]);
      }
    }
  }

  // ---- epilogue: add down_b, scatter rows to tokens ----
  int tokr[8];
  bool vr[8];
#pragma unroll
  for (int j = 0; j < 8; ++j) {
    int m = j + 8 * half;
    int p = tilebase + m;
    vr[j] = p < cnt;
    tokr[j] = vr[j] ? idx[p] : 0;
  }
#pragma unroll
  for (int t = 0; t < 16; ++t) {
    int col = 256 * w + 16 * t + nlo;
    float db = down_b[col];
#pragma unroll
    for (int j = 0; j < 8; ++j) {
      if (vr[j]) out[(size_t)tokr[j] * HDIM + col] = acc[t][j] + db;
    }
  }
}

// ---------------- host launcher ----------------
extern "C" void kernel_launch(void* const* d_in, const int* in_sizes, int n_in,
                              void* d_out, int out_size, void* d_ws, size_t ws_size,
                              hipStream_t stream) {
  const float* hs     = (const float*)d_in[0];
  const int*   mask   = (const int*)d_in[1];
  const float* lnw    = (const float*)d_in[2];
  const float* gate_w = (const float*)d_in[3];
  const float* gate_b = (const float*)d_in[4];
  const float* up_w   = (const float*)d_in[5];
  const float* up_b   = (const float*)d_in[6];
  const float* down_w = (const float*)d_in[7];
  const float* down_b = (const float*)d_in[8];
  float* out = (float*)d_out;

  char* ws = (char*)d_ws;
  int* cnt = (int*)(ws + OFF_CNT);
  int* idx = (int*)(ws + OFF_IDX);
  unsigned int* wg_hi = (unsigned int*)(ws + OFF_WGH);
  unsigned int* wg_lo = (unsigned int*)(ws + OFF_WGL);
  unsigned int* wu_hi = (unsigned int*)(ws + OFF_WUH);
  unsigned int* wu_lo = (unsigned int*)(ws + OFF_WUL);
  unsigned int* wd_hi = (unsigned int*)(ws + OFF_WDH);
  unsigned int* wd_lo = (unsigned int*)(ws + OFF_WDL);

  init_out_kernel<<<NTOK, 256, 0, stream>>>(mask, out);
  compact_kernel<<<1, 1024, 0, stream>>>(mask, idx, cnt);

  int cvt_blocks = (W_DWORDS + 255) / 256;
  convert_swz_kernel<<<cvt_blocks, 256, 0, stream>>>(gate_w, wg_hi, wg_lo, HDIM, IDIM);
  convert_swz_kernel<<<cvt_blocks, 256, 0, stream>>>(up_w, wu_hi, wu_lo, HDIM, IDIM);
  convert_swz_kernel<<<cvt_blocks, 256, 0, stream>>>(down_w, wd_hi, wd_lo, IDIM, HDIM);

  mlp_main_kernel<<<NTOK / 16, 256, 0, stream>>>(hs, lnw, gate_b, up_b, down_b, idx,
                                                 cnt, wg_hi, wg_lo, wu_hi, wu_lo,
                                                 wd_hi, wd_lo, out);
}
